// UniCMP_61787399520924
// MI455X (gfx1250) — compile-verified, matrix operand order checked by reading, not run on hardware
//
#include <hip/hip_runtime.h>
#include <math.h>

// ---- problem constants ----
#define N0v   131072
#define N1v   32768
#define N2v   4096
#define KNv   8
#define Fv    300
#define Hv    512
#define NHv   4
#define DHv   128
#define Cv    23

typedef __attribute__((ext_vector_type(16))) __bf16          v16bf;
typedef __attribute__((ext_vector_type(8)))  float           v8f;
typedef __attribute__((ext_vector_type(16))) unsigned short  v16us;
typedef __attribute__((ext_vector_type(8)))  unsigned short  v8us;

__device__ __forceinline__ unsigned short f2bfu(float x) {
    unsigned u = __builtin_bit_cast(unsigned, x);
    u += 0x7FFFu + ((u >> 16) & 1u);               // round-to-nearest-even
    return (unsigned short)(u >> 16);
}
__device__ __forceinline__ float bfu2f(unsigned short h) {
    unsigned u = ((unsigned)h) << 16;
    return __builtin_bit_cast(float, u);
}

// =====================================================================
// WMMA GEMM, all-bf16, fully padded operands (no in-loop conditionals):
//   C[M,N] (+)= A[M,Kd] * W[Kd,N] (+bias)
// Requirements (guaranteed by host): Kd%32==0, lda%8==0, ldw%32==0,
// M%128==0; zero-padded K/N tails in A and W.
// One wave computes a 16x32 C strip (2 WMMAs/k-step, A reused);
// 8 waves share a 32x32 B tile staged via LDS ([n][k] transposed).
// =====================================================================
__global__ __launch_bounds__(256)
void wmma_gemm_kernel(const unsigned short* __restrict__ A, int lda,
                      const int* __restrict__ rowIdx, int idxStride,
                      const unsigned short* __restrict__ W, int ldw,
                      const float* __restrict__ bias,
                      float* __restrict__ C, int ldc,
                      int M, int Kd, int N, int accFlag)
{
    __shared__ __align__(16) unsigned short bsm[32 * 32];   // bsm[n*32 + k]
    const int tid   = threadIdx.x;
    const int wave  = tid >> 5;
    const int lane  = tid & 31;
    const int tileM = (blockIdx.y * 8 + wave) * 16;
    const int tileN = blockIdx.x * 32;
    const int halfSel = lane >> 4;                 // 0 or 1
    const int lm      = lane & 15;
    const int kbase   = halfSel * 8;

    const int rowA = tileM + lm;                   // rowA < M always
    const int r = rowIdx ? rowIdx[(long)rowA * idxStride] : rowA;
    const unsigned short* Ap = A + (long)r * lda + kbase;

    const int cn0 = tileN + lm;
    const int cn1 = tileN + 16 + lm;

    v8f acc0, acc1;
#pragma unroll
    for (int rr = 0; rr < 8; ++rr) { acc0[rr] = 0.f; acc1[rr] = 0.f; }
    if (accFlag) {
#pragma unroll
        for (int rr = 0; rr < 8; ++rr) {
            int m = tileM + rr + 8 * halfSel;
            float v0 = C[(long)m * ldc + min(cn0, N - 1)];
            float v1 = C[(long)m * ldc + min(cn1, N - 1)];
            acc0[rr] = (cn0 < N) ? v0 : 0.f;
            acc1[rr] = (cn1 < N) ? v1 : 0.f;
        }
    }

    // cooperative staging: 1024 elems, 4/thread; n fast-varying -> coalesced
    const int sn  = tid & 31;
    const int sk0 = tid >> 5;                      // k = sk0 + e*8
    const unsigned short* Wp = W + tileN + sn;

    for (int k0 = 0; k0 < Kd; k0 += 32) {
        __syncthreads();
#pragma unroll
        for (int e = 0; e < 4; ++e) {
            int k = sk0 + e * 8;
            bsm[sn * 32 + k] = Wp[(long)(k0 + k) * ldw];
        }
        __syncthreads();

        // speculative next-slice prefetch (gfx1250 global_prefetch)
        __builtin_prefetch(Ap + k0 + 32, 0, 1);
        __builtin_prefetch(Wp + (long)(k0 + 32 + sk0) * ldw, 0, 1);

        v16us au, bu0, bu1;
        {
            v8us a0 = *reinterpret_cast<const v8us*>(Ap + k0);
            v8us a1 = *reinterpret_cast<const v8us*>(Ap + k0 + 16);
#pragma unroll
            for (int i = 0; i < 8; ++i) { au[i] = a0[i]; au[i + 8] = a1[i]; }
        }
        {
            v8us b0 = *reinterpret_cast<const v8us*>(&bsm[lm * 32 + kbase]);
            v8us b1 = *reinterpret_cast<const v8us*>(&bsm[lm * 32 + kbase + 16]);
            v8us c0 = *reinterpret_cast<const v8us*>(&bsm[(lm + 16) * 32 + kbase]);
            v8us c1 = *reinterpret_cast<const v8us*>(&bsm[(lm + 16) * 32 + kbase + 16]);
#pragma unroll
            for (int i = 0; i < 8; ++i) {
                bu0[i] = b0[i]; bu0[i + 8] = b1[i];
                bu1[i] = c0[i]; bu1[i + 8] = c1[i];
            }
        }
        acc0 = __builtin_amdgcn_wmma_f32_16x16x32_bf16(
                   false, __builtin_bit_cast(v16bf, au),
                   false, __builtin_bit_cast(v16bf, bu0),
                   (short)0, acc0, false, false);
        acc1 = __builtin_amdgcn_wmma_f32_16x16x32_bf16(
                   false, __builtin_bit_cast(v16bf, au),
                   false, __builtin_bit_cast(v16bf, bu1),
                   (short)0, acc1, false, false);
    }

#pragma unroll
    for (int rr = 0; rr < 8; ++rr) {
        int m = tileM + rr + 8 * halfSel;
        long rowoff = (long)m * ldc;
        if (cn0 < N) {
            float v = acc0[rr];
            if (bias) v += bias[cn0];
            C[rowoff + cn0] = v;
        }
        if (cn1 < N) {
            float v = acc1[rr];
            if (bias) v += bias[cn1];
            C[rowoff + cn1] = v;
        }
    }
}

// =====================================================================
// LayerNorm with fused activation (0=none,1=relu,2=elu); dual f32/bf16 out
// =====================================================================
__global__ __launch_bounds__(256)
void layernorm_kernel(const float* __restrict__ x, const float* __restrict__ g,
                      const float* __restrict__ b,
                      float* __restrict__ outf, unsigned short* __restrict__ outb,
                      int Wd, int act)
{
    __shared__ float red[256];
    __shared__ float stat[2];
    const long row = blockIdx.x;
    const float* xr = x + row * Wd;
    const int tid = threadIdx.x;

    float s = 0.f, s2 = 0.f;
    for (int c = tid; c < Wd; c += 256) { float v = xr[c]; s += v; s2 += v * v; }
    red[tid] = s; __syncthreads();
    for (int o = 128; o > 0; o >>= 1) { if (tid < o) red[tid] += red[tid + o]; __syncthreads(); }
    if (tid == 0) stat[0] = red[0] / Wd;
    __syncthreads();
    red[tid] = s2; __syncthreads();
    for (int o = 128; o > 0; o >>= 1) { if (tid < o) red[tid] += red[tid + o]; __syncthreads(); }
    if (tid == 0) {
        float m = stat[0];
        stat[1] = rsqrtf(red[0] / Wd - m * m + 1e-12f);
    }
    __syncthreads();
    const float mean = stat[0], inv = stat[1];
    for (int c = tid; c < Wd; c += 256) {
        float v = g[c] * (xr[c] - mean) * inv + b[c];
        if (act == 1)      v = fmaxf(v, 0.f);
        else if (act == 2) v = (v > 0.f) ? v : (expf(v) - 1.f);
        if (outf) outf[row * Wd + c] = v;
        if (outb) outb[row * Wd + c] = f2bfu(v);
    }
}

// ---- weight conversion (zero-padded bf16) ----
// dst (Kp,Np) = src (R,Cc) padded with zeros
__global__ void cvtw_pad_kernel(const float* __restrict__ src,
                                unsigned short* __restrict__ dst,
                                int R, int Cc, int Kp, int Np)
{
    int i = blockIdx.x * blockDim.x + threadIdx.x;
    if (i >= Kp * Np) return;
    int r = i / Np, c = i % Np;
    dst[i] = (r < R && c < Cc) ? f2bfu(src[(long)r * Cc + c]) : (unsigned short)0;
}
// dst (Kp,Np) = src^T, src (R,Cc): dst[k][n] = src[n][k]
__global__ void cvtwT_pad_kernel(const float* __restrict__ src,
                                 unsigned short* __restrict__ dst,
                                 int R, int Cc, int Kp, int Np)
{
    int i = blockIdx.x * blockDim.x + threadIdx.x;
    if (i >= Kp * Np) return;
    int k = i / Np, n = i % Np;
    dst[i] = (k < Cc && n < R) ? f2bfu(src[(long)n * Cc + k]) : (unsigned short)0;
}
// activation pad-convert: src (rows,cols) f32 -> dst bf16 ld=ldd, zero tail
__global__ void cvt_pad_kernel(const float* __restrict__ src,
                               unsigned short* __restrict__ dst,
                               int rows, int cols, int ldd)
{
    int i = blockIdx.x * blockDim.x + threadIdx.x;
    if (i >= rows * ldd) return;
    int r = i / ldd, c = i % ldd;
    dst[i] = (c < cols) ? f2bfu(src[(long)r * cols + c]) : (unsigned short)0;
}

__global__ void zero_f_kernel(float* __restrict__ p, int n)
{
    int i = blockIdx.x * blockDim.x + threadIdx.x;
    if (i < n) p[i] = 0.f;
}
__global__ void zero_u_kernel(unsigned short* __restrict__ p, int n)
{
    int i = blockIdx.x * blockDim.x + threadIdx.x;
    if (i < n) p[i] = 0;
}

// feat = [label_embed[labels] , input_feats], bf16, ld=ldd, zero tail
__global__ void embed_concat_kernel(const float* __restrict__ feats,
                                    const int* __restrict__ labels,
                                    const float* __restrict__ emb,
                                    unsigned short* __restrict__ out,
                                    int total, int ldd)
{
    int idx = blockIdx.x * blockDim.x + threadIdx.x;
    if (idx >= total) return;
    int row = idx / ldd;
    int col = idx % ldd;
    float v = 0.f;
    if (col < Fv)           v = emb[(long)labels[row] * Fv + col];
    else if (col < 2 * Fv)  v = feats[(long)row * Fv + (col - Fv)];
    out[idx] = f2bfu(v);
}

__global__ void lstm_gate_kernel(const float* __restrict__ z,
                                 float* __restrict__ c,
                                 unsigned short* __restrict__ hb,
                                 int n, int d, int ldp)
{
    int idx = blockIdx.x * blockDim.x + threadIdx.x;
    if (idx >= n * d) return;
    int row = idx / d, col = idx % d;
    const float* zr = z + (long)row * 4 * d;
    float ig = zr[col], fg = zr[d + col], gg = zr[2 * d + col], og = zr[3 * d + col];
    float si = 1.f / (1.f + expf(-ig));
    float sf = 1.f / (1.f + expf(-fg));
    float so = 1.f / (1.f + expf(-og));
    float cn = sf * c[idx] + si * tanhf(gg);
    c[idx] = cn;
    hb[(long)row * ldp + col] = f2bfu(so * tanhf(cn));
}

__global__ void scatter_deg_kernel(const int* __restrict__ nb, int cnt,
                                   float* __restrict__ deg)
{
    int i = blockIdx.x * blockDim.x + threadIdx.x;
    if (i < cnt) atomicAdd(&deg[nb[i]], 1.0f);
}

__global__ void agg_kernel(const unsigned short* __restrict__ hb,
                           const int* __restrict__ nb, const float* __restrict__ deg,
                           unsigned short* __restrict__ outb,
                           int n_dst, int d, int ldp, float scale)
{
    int idx = blockIdx.x * blockDim.x + threadIdx.x;
    if (idx >= n_dst * ldp) return;
    int row = idx / ldp, col = idx % ldp;
    if (col >= d) { outb[idx] = 0; return; }       // keep K padding zero
    float s = 0.f;
#pragma unroll
    for (int kk = 0; kk < KNv; ++kk) {
        int src = nb[row * KNv + kk];
        s += bfu2f(hb[(long)src * ldp + col]) * rsqrtf(fmaxf(deg[src], 1.f));
    }
    outb[idx] = f2bfu(s * scale);
}

__global__ void gat_el_kernel(const float* __restrict__ fsrc,
                              const float* __restrict__ av,
                              float* __restrict__ out, int n)
{
    int idx = blockIdx.x * blockDim.x + threadIdx.x;
    if (idx >= n * NHv) return;
    int i = idx / NHv, hd = idx % NHv;
    const float* f = fsrc + (long)i * Hv + hd * DHv;
    const float* a = av + hd * DHv;
    float s = 0.f;
    for (int d2 = 0; d2 < DHv; ++d2) s += f[d2] * a[d2];
    out[idx] = s;
}

__global__ __launch_bounds__(128)
void gat_attn_kernel(const float* __restrict__ fsrc, const float* __restrict__ el,
                     const float* __restrict__ er, const int* __restrict__ nb,
                     const float* __restrict__ bgat, float* __restrict__ out, int n_dst)
{
    __shared__ float e_sm[KNv * NHv];
    __shared__ float a_sm[KNv * NHv];
    __shared__ int   idx_sm[KNv];
    const int i = blockIdx.x;
    const int tid = threadIdx.x;
    if (tid < KNv) idx_sm[tid] = nb[i * KNv + tid];
    __syncthreads();
    if (tid < KNv * NHv) {
        int kk = tid / NHv, hd = tid % NHv;
        float e = el[(long)idx_sm[kk] * NHv + hd] + er[(long)i * NHv + hd];
        e_sm[tid] = (e > 0.f) ? e : 0.2f * e;      // leaky_relu(0.2)
    }
    __syncthreads();
    if (tid < NHv) {
        float mx = -1e30f;
        for (int kk = 0; kk < KNv; ++kk) mx = fmaxf(mx, e_sm[kk * NHv + tid]);
        float sum = 0.f;
        for (int kk = 0; kk < KNv; ++kk) {
            float ex = expf(e_sm[kk * NHv + tid] - mx);
            a_sm[kk * NHv + tid] = ex; sum += ex;
        }
        float inv = 1.f / sum;
        for (int kk = 0; kk < KNv; ++kk) a_sm[kk * NHv + tid] *= inv;
    }
    __syncthreads();
    for (int c = tid; c < Hv; c += 128) {
        int hd = c / DHv;
        float acc = bgat[c];
#pragma unroll
        for (int kk = 0; kk < KNv; ++kk)
            acc += a_sm[kk * NHv + hd] * fsrc[(long)idx_sm[kk] * Hv + c];
        out[(long)i * Hv + c] = acc;
    }
}

__global__ void copy_branch_kernel(const unsigned short* __restrict__ src,
                                   unsigned short* __restrict__ dst,
                                   int n, int B, int b)
{
    int idx = blockIdx.x * blockDim.x + threadIdx.x;
    if (idx >= n * Hv) return;
    int row = idx / Hv, col = idx % Hv;
    dst[((long)row * B + b) * Hv + col] = src[(long)row * Hv + col];
}

// Branch attention: only query token 0 needed; one wave per (node, head)
template <int B>
__global__ __launch_bounds__(128)
void branch_attn_kernel(const float* __restrict__ q, const float* __restrict__ k,
                        const float* __restrict__ v, float* __restrict__ out)
{
    const int node = blockIdx.x;
    const int head = threadIdx.x >> 5;
    const int lane = threadIdx.x & 31;
    const float scale = 0.0883883476483f;          // 1/sqrt(128)
    const long base = ((long)node * B) * Hv + head * DHv + lane * 4;

    const float4 q0 = *reinterpret_cast<const float4*>(q + base);
    float s[B];
#pragma unroll
    for (int bb = 0; bb < B; ++bb) {
        const float4 kb = *reinterpret_cast<const float4*>(k + base + (long)bb * Hv);
        float p = q0.x * kb.x + q0.y * kb.y + q0.z * kb.z + q0.w * kb.w;
#pragma unroll
        for (int off = 16; off > 0; off >>= 1) p += __shfl_xor(p, off, 32);
        s[bb] = p * scale;
    }
    float mx = s[0];
#pragma unroll
    for (int bb = 1; bb < B; ++bb) mx = fmaxf(mx, s[bb]);
    float sum = 0.f;
#pragma unroll
    for (int bb = 0; bb < B; ++bb) { s[bb] = expf(s[bb] - mx); sum += s[bb]; }
    const float inv = 1.f / sum;
    float4 ctx = make_float4(0.f, 0.f, 0.f, 0.f);
#pragma unroll
    for (int bb = 0; bb < B; ++bb) {
        const float4 vb = *reinterpret_cast<const float4*>(v + base + (long)bb * Hv);
        float a = s[bb] * inv;
        ctx.x += a * vb.x; ctx.y += a * vb.y; ctx.z += a * vb.z; ctx.w += a * vb.w;
    }
    *reinterpret_cast<float4*>(out + (long)node * Hv + head * DHv + lane * 4) = ctx;
}

// =====================================================================
// Host side
// =====================================================================
static inline int cdiv(int a, int b) { return (a + b - 1) / b; }
static inline int pad32(int x) { return (x + 31) & ~31; }

static void gemm(hipStream_t s, const unsigned short* A, int lda,
                 const int* ridx, int istr,
                 const unsigned short* W, int ldw, const float* bias,
                 float* C, int ldc, int M, int Kd, int N, int acc)
{
    dim3 grid(cdiv(N, 32), cdiv(M, 128));
    wmma_gemm_kernel<<<grid, 256, 0, s>>>(A, lda, ridx, istr, W, ldw,
                                          bias, C, ldc, M, Kd, N, acc);
}

static void ln(hipStream_t s, const float* x, const float* g, const float* b,
               float* outf, unsigned short* outb, int M, int Wd, int act)
{
    layernorm_kernel<<<M, 256, 0, s>>>(x, g, b, outf, outb, Wd, act);
}

static void cvtW(hipStream_t s, const float* src, unsigned short* dst,
                 int R, int Cc, int Kp, int Np)
{
    cvtw_pad_kernel<<<cdiv(Kp * Np, 256), 256, 0, s>>>(src, dst, R, Cc, Kp, Np);
}
static void cvtWT(hipStream_t s, const float* src, unsigned short* dst,
                  int R, int Cc, int Kp, int Np)
{
    cvtwT_pad_kernel<<<cdiv(Kp * Np, 256), 256, 0, s>>>(src, dst, R, Cc, Kp, Np);
}

struct LayerP {   // f32 params straight from d_in
    const float *Wgat,*Wgc,*Whh,*Wih,*Wk,*Wneigh,*Wq,*Wself,*Wsk,*Wv;
    const float *al,*ar,*bgat,*bgc,*bhh,*bih,*bk,*bq,*bsage,*bsk,*bv,*ln_b,*ln_g;
};
struct LayerW {   // bf16 padded weights
    unsigned short *Wsk,*Wself,*Wneigh,*Wgc,*Wgat,*WihT,*WhhT,*Wq,*Wk,*Wv;
};
struct Scratch {
    float *h1f,*h2f,*h3f,*h4f,*hscf,*zg,*lstm_c,*fsrc,*el,*er,*deg;
    float *q,*k,*v,*attn;
    unsigned short *h1b,*h2b,*h3b,*h4b,*hscb,*lstm_hb,*aggb,*xstack;
};

static void run_layer(hipStream_t s, const LayerP& p, const LayerW& wt,
                      const unsigned short* hb, int in_f, int ldp,
                      int n_src, int n_dst, const int* nb,
                      const unsigned short* featDstB, int featLd,
                      const unsigned short* scWb, const float* scb,
                      const float* scg, const float* scbe,
                      const Scratch& w, unsigned short* hout_b)
{
    const int G  = 4 * in_f;           // gate width (1200 / 2048)
    const int Gp = pad32(G);           // padded weight ldw (1216 / 2048)

    // --- branch 1: skip MLP ---
    gemm(s, hb, ldp, nullptr, 0, wt.Wsk, Hv, p.bsk, w.h1f, Hv, n_dst, ldp, Hv, 0);
    ln(s, w.h1f, p.ln_g + 0 * Hv, p.ln_b + 0 * Hv, nullptr, w.h1b, n_dst, Hv, 2);

    // --- branch 2: LSTM over K neighbors (gather fused into GEMM A) ---
    const int nd = n_dst * in_f;
    zero_u_kernel<<<cdiv(n_dst * ldp, 256), 256, 0, s>>>(w.lstm_hb, n_dst * ldp);
    zero_f_kernel<<<cdiv(nd, 256), 256, 0, s>>>(w.lstm_c, nd);
    for (int t = 0; t < KNv; ++t) {
        gemm(s, hb, ldp, nb + t, KNv, wt.WihT, Gp, p.bih, w.zg, G, n_dst, ldp, G, 0);
        gemm(s, w.lstm_hb, ldp, nullptr, 0, wt.WhhT, Gp, p.bhh, w.zg, G, n_dst, ldp, G, 1);
        lstm_gate_kernel<<<cdiv(nd, 256), 256, 0, s>>>(w.zg, w.lstm_c, w.lstm_hb,
                                                       n_dst, in_f, ldp);
    }
    gemm(s, hb, ldp, nullptr, 0, wt.Wself, Hv, p.bsage, w.h2f, Hv, n_dst, ldp, Hv, 0);
    gemm(s, w.lstm_hb, ldp, nullptr, 0, wt.Wneigh, Hv, nullptr, w.h2f, Hv, n_dst, ldp, Hv, 1);
    ln(s, w.h2f, p.ln_g + 1 * Hv, p.ln_b + 1 * Hv, nullptr, w.h2b, n_dst, Hv, 2);

    // --- branch 3: GCN-style normalized mean ---
    zero_f_kernel<<<cdiv(n_src, 256), 256, 0, s>>>(w.deg, n_src);
    scatter_deg_kernel<<<cdiv(n_dst * KNv, 256), 256, 0, s>>>(nb, n_dst * KNv, w.deg);
    agg_kernel<<<cdiv(n_dst * ldp, 256), 256, 0, s>>>(hb, nb, w.deg, w.aggb,
                                                      n_dst, in_f, ldp,
                                                      0.3535533905932738f); // 1/sqrt(8)
    gemm(s, w.aggb, ldp, nullptr, 0, wt.Wgc, Hv, p.bgc, w.h3f, Hv, n_dst, ldp, Hv, 0);
    ln(s, w.h3f, p.ln_g + 2 * Hv, p.ln_b + 2 * Hv, nullptr, w.h3b, n_dst, Hv, 2);

    // --- branch 4: GAT ---
    gemm(s, hb, ldp, nullptr, 0, wt.Wgat, Hv, nullptr, w.fsrc, Hv, n_src, ldp, Hv, 0);
    gat_el_kernel<<<cdiv(n_src * NHv, 256), 256, 0, s>>>(w.fsrc, p.al, w.el, n_src);
    gat_el_kernel<<<cdiv(n_dst * NHv, 256), 256, 0, s>>>(w.fsrc, p.ar, w.er, n_dst);
    gat_attn_kernel<<<n_dst, 128, 0, s>>>(w.fsrc, w.el, w.er, nb, p.bgat, w.h4f, n_dst);
    ln(s, w.h4f, p.ln_g + 3 * Hv, p.ln_b + 3 * Hv, nullptr, w.h4b, n_dst, Hv, 2);

    // --- branch 5 (last layer only): skip connection from original feature ---
    const int B = scWb ? 5 : 4;
    if (scWb) {
        gemm(s, featDstB, featLd, nullptr, 0, scWb, Hv, scb, w.hscf, Hv,
             n_dst, featLd, Hv, 0);
        ln(s, w.hscf, scg, scbe, nullptr, w.hscb, n_dst, Hv, 0);
    }

    // --- stack + Q/K/V + branch attention (token 0) ---
    const unsigned short* branches[5] = { w.h1b, w.h2b, w.h3b, w.h4b, w.hscb };
    for (int bb = 0; bb < B; ++bb)
        copy_branch_kernel<<<cdiv(n_dst * Hv, 256), 256, 0, s>>>(branches[bb], w.xstack,
                                                                 n_dst, B, bb);
    const int Mqkv = n_dst * B;        // 131072 or 20480 (both %128==0)
    gemm(s, w.xstack, Hv, nullptr, 0, wt.Wq, Hv, p.bq, w.q, Hv, Mqkv, Hv, Hv, 0);
    gemm(s, w.xstack, Hv, nullptr, 0, wt.Wk, Hv, p.bk, w.k, Hv, Mqkv, Hv, Hv, 0);
    gemm(s, w.xstack, Hv, nullptr, 0, wt.Wv, Hv, p.bv, w.v, Hv, Mqkv, Hv, Hv, 0);
    if (B == 4) branch_attn_kernel<4><<<n_dst, 128, 0, s>>>(w.q, w.k, w.v, w.attn);
    else        branch_attn_kernel<5><<<n_dst, 128, 0, s>>>(w.q, w.k, w.v, w.attn);
    ln(s, w.attn, p.ln_g + 4 * Hv, p.ln_b + 4 * Hv, nullptr, hout_b, n_dst, Hv, 0);
}

extern "C" void kernel_launch(void* const* d_in, const int* in_sizes, int n_in,
                              void* d_out, int out_size, void* d_ws, size_t ws_size,
                              hipStream_t stream)
{
    (void)in_sizes; (void)n_in; (void)out_size; (void)ws_size;

    auto f = [&](int i) { return (const float*)d_in[i]; };
    const float* input_feats  = f(0);
    const int*   input_labels = (const int*)d_in[1];
    const int*   neigh0       = (const int*)d_in[2];
    const int*   neigh1       = (const int*)d_in[3];

    // params pytree (JAX dict-key-sorted flatten order)
    const float *fm_W1 = f(4), *fm_W2 = f(5), *fm_b1 = f(6), *fm_b2 = f(7),
                *fm_be1 = f(8), *fm_g1 = f(9);
    const float *hd_W1 = f(10), *hd_W2 = f(11), *hd_b1 = f(12), *hd_b2 = f(13),
                *hd_be = f(14), *hd_g = f(15);
    const float *emb = f(16);
    auto loadLayer = [&](int b) {
        LayerP p;
        p.Wgat = f(b+0);  p.Wgc = f(b+1);  p.Whh = f(b+2);   p.Wih = f(b+3);
        p.Wk = f(b+4);    p.Wneigh = f(b+5); p.Wq = f(b+6);  p.Wself = f(b+7);
        p.Wsk = f(b+8);   p.Wv = f(b+9);   p.al = f(b+10);   p.ar = f(b+11);
        p.bgat = f(b+12); p.bgc = f(b+13); p.bhh = f(b+14);  p.bih = f(b+15);
        p.bk = f(b+16);   p.bq = f(b+17);  p.bsage = f(b+18); p.bsk = f(b+19);
        p.bv = f(b+20);   p.ln_b = f(b+21); p.ln_g = f(b+22);
        return p;
    };
    LayerP L0 = loadLayer(17);
    LayerP L1 = loadLayer(40);
    const float *sc_W = f(63), *sc_b = f(64), *sc_be = f(65), *sc_g = f(66);

    // ---- carve workspace ----
    char* base = (char*)d_ws;
    size_t off = 0;
    auto allocF = [&](size_t n) { float* p = (float*)(base + off);
                                  off += ((n * 4) + 255) & ~(size_t)255; return p; };
    auto allocU = [&](size_t n) { unsigned short* p = (unsigned short*)(base + off);
                                  off += ((n * 2) + 255) & ~(size_t)255; return p; };

    const int ldF  = pad32(Fv);        // 320
    const int ld2F = pad32(2 * Fv);    // 608
    const int G0p  = pad32(4 * Fv);    // 1216
    // f32 scratch
    float* z1f      = allocF((size_t)N0v * Hv);
    float* featuref = allocF((size_t)N0v * Fv);
    Scratch w;
    w.h1f   = allocF((size_t)N1v * Hv);
    w.h2f   = allocF((size_t)N1v * Hv);
    w.h3f   = allocF((size_t)N1v * Hv);
    w.h4f   = allocF((size_t)N1v * Hv);
    w.hscf  = allocF((size_t)N2v * Hv);
    w.zg    = allocF((size_t)N1v * 4 * Hv);
    w.lstm_c= allocF((size_t)N1v * Hv);
    w.fsrc  = allocF((size_t)N0v * Hv);
    w.el    = allocF((size_t)N0v * NHv);
    w.er    = allocF((size_t)N1v * NHv);
    w.deg   = allocF((size_t)N0v);
    // bf16 scratch (all K-padded / zero-tailed)
    unsigned short* feat_b    = allocU((size_t)N0v * ld2F);
    unsigned short* z1_b      = allocU((size_t)N0v * Hv);
    unsigned short* feature_b = allocU((size_t)N0v * ldF);
    unsigned short* hL1_b     = allocU((size_t)N1v * Hv);
    unsigned short* hL2_b     = allocU((size_t)N2v * Hv);
    w.h1b     = allocU((size_t)N1v * Hv);
    w.h2b     = allocU((size_t)N1v * Hv);
    w.h3b     = allocU((size_t)N1v * Hv);
    w.h4b     = allocU((size_t)N1v * Hv);
    w.hscb    = allocU((size_t)N2v * Hv);
    w.lstm_hb = allocU((size_t)N1v * Hv);
    w.aggb    = allocU((size_t)N1v * Hv);
    w.xstack  = allocU((size_t)N1v * 4 * Hv);      // >= N2*5*H too
    // bf16 weights, (pad32(K) x pad32(N))
    unsigned short* wb_fmW1 = allocU((size_t)ld2F * Hv);
    unsigned short* wb_fmW2 = allocU((size_t)Hv * ldF);
    unsigned short* wb_hdW1 = allocU((size_t)Hv * Hv);
    unsigned short* wb_hdW2 = allocU((size_t)Hv * 32);
    unsigned short* wb_scW  = allocU((size_t)ldF * Hv);
    LayerW wt0, wt1;
    auto allocLW = [&](LayerW& lw, int in_f) {
        const int kp = pad32(in_f), gp = pad32(4 * in_f);
        lw.Wsk    = allocU((size_t)kp * Hv);
        lw.Wself  = allocU((size_t)kp * Hv);
        lw.Wneigh = allocU((size_t)kp * Hv);
        lw.Wgc    = allocU((size_t)kp * Hv);
        lw.Wgat   = allocU((size_t)kp * Hv);
        lw.WihT   = allocU((size_t)kp * gp);
        lw.WhhT   = allocU((size_t)kp * gp);
        lw.Wq     = allocU((size_t)Hv * Hv);
        lw.Wk     = allocU((size_t)Hv * Hv);
        lw.Wv     = allocU((size_t)Hv * Hv);
    };
    allocLW(wt0, Fv);
    allocLW(wt1, Hv);
    // f32 aliases of dead buffers for Q/K/V & attn temp
    w.q    = z1f;        // dead after feature GEMM
    w.k    = w.fsrc;     // dead after GAT branch
    w.v    = w.zg;       // dead after LSTM
    w.attn = w.h1f;      // dead after h1 LN

    // ---- weight conversion (bf16, zero-padded; Wih/Whh pre-transposed) ----
    cvtW(stream, fm_W1, wb_fmW1, 2 * Fv, Hv, ld2F, Hv);
    cvtW(stream, fm_W2, wb_fmW2, Hv, Fv, Hv, ldF);
    cvtW(stream, hd_W1, wb_hdW1, Hv, Hv, Hv, Hv);
    cvtW(stream, hd_W2, wb_hdW2, Hv, Cv, Hv, 32);
    cvtW(stream, sc_W,  wb_scW,  Fv, Hv, ldF, Hv);
    auto cvtLayer = [&](const LayerP& p, LayerW& lw, int in_f) {
        const int kp = pad32(in_f), gp = pad32(4 * in_f);
        cvtW(stream, p.Wsk,    lw.Wsk,    in_f, Hv, kp, Hv);
        cvtW(stream, p.Wself,  lw.Wself,  in_f, Hv, kp, Hv);
        cvtW(stream, p.Wneigh, lw.Wneigh, in_f, Hv, kp, Hv);
        cvtW(stream, p.Wgc,    lw.Wgc,    in_f, Hv, kp, Hv);
        cvtW(stream, p.Wgat,   lw.Wgat,   in_f, Hv, kp, Hv);
        cvtWT(stream, p.Wih,   lw.WihT,   4 * in_f, in_f, kp, gp);  // (4d,d)->(d,4d)
        cvtWT(stream, p.Whh,   lw.WhhT,   4 * in_f, in_f, kp, gp);
        cvtW(stream, p.Wq,     lw.Wq,     Hv, Hv, Hv, Hv);
        cvtW(stream, p.Wk,     lw.Wk,     Hv, Hv, Hv, Hv);
        cvtW(stream, p.Wv,     lw.Wv,     Hv, Hv, Hv, Hv);
    };
    cvtLayer(L0, wt0, Fv);
    cvtLayer(L1, wt1, Hv);

    // ---- 1. label embedding + concat (bf16, zero tail) ----
    embed_concat_kernel<<<cdiv(N0v * ld2F, 256), 256, 0, stream>>>(
        input_feats, input_labels, emb, feat_b, N0v * ld2F, ld2F);

    // ---- 2. feature MLP ----
    gemm(stream, feat_b, ld2F, nullptr, 0, wb_fmW1, Hv, fm_b1, z1f, Hv,
         N0v, ld2F, Hv, 0);
    ln(stream, z1f, fm_g1, fm_be1, nullptr, z1_b, N0v, Hv, 1);
    gemm(stream, z1_b, Hv, nullptr, 0, wb_fmW2, ldF, fm_b2, featuref, Fv,
         N0v, Hv, Fv, 0);
    cvt_pad_kernel<<<cdiv(N0v * ldF, 256), 256, 0, stream>>>(featuref, feature_b,
                                                             N0v, Fv, ldF);

    // ---- 3. GNN layers ----
    run_layer(stream, L0, wt0, feature_b, Fv, ldF, N0v, N1v, neigh0,
              nullptr, 0, nullptr, nullptr, nullptr, nullptr, w, hL1_b);
    run_layer(stream, L1, wt1, hL1_b, Hv, Hv, N1v, N2v, neigh1,
              feature_b, ldF, wb_scW, sc_b, sc_g, sc_be, w, hL2_b);

    // ---- 4. head ----
    gemm(stream, hL2_b, Hv, nullptr, 0, wb_hdW1, Hv, hd_b1, w.h1f, Hv, N2v, Hv, Hv, 0);
    ln(stream, w.h1f, hd_g, hd_be, nullptr, w.h1b, N2v, Hv, 1);
    gemm(stream, w.h1b, Hv, nullptr, 0, wb_hdW2, 32, hd_b2,
         (float*)d_out, Cv, N2v, Hv, Cv, 0);
}
